// HAGMoE_90013924590140
// MI455X (gfx1250) — matrix operand order repaired and checked by reading, not compiled, not run on hardware
//
#include <hip/hip_runtime.h>
#include <math.h>

// ---------------- problem constants ----------------
#define BB   2048
#define SS   128
#define HH   768
#define II   3072
#define GG   3
#define EE   8
#define LL   3
#define NEXP (GG*EE)   // 24

// ---------------- bf16 helpers ----------------
__device__ __forceinline__ unsigned short f2bf(float f) {
    unsigned u = __builtin_bit_cast(unsigned, f);
    unsigned r = u + 0x7FFFu + ((u >> 16) & 1u);   // round-to-nearest-even
    return (unsigned short)(r >> 16);
}

typedef __attribute__((ext_vector_type(16))) __bf16 v16bf;
typedef __attribute__((ext_vector_type(8)))  float  v8f;

union Frag { v16bf v; unsigned u[8]; };

// ---------------- block reductions ----------------
__device__ __forceinline__ float block_sum(float v, float* red, int t) {
    red[t] = v; __syncthreads();
    #pragma unroll
    for (int o = 128; o > 0; o >>= 1) {
        if (t < o) red[t] += red[t + o];
        __syncthreads();
    }
    float r = red[0]; __syncthreads();
    return r;
}
__device__ __forceinline__ float block_max(float v, float* red, int t) {
    red[t] = v; __syncthreads();
    #pragma unroll
    for (int o = 128; o > 0; o >>= 1) {
        if (t < o) red[t] = fmaxf(red[t], red[t + o]);
        __syncthreads();
    }
    float r = red[0]; __syncthreads();
    return r;
}

// =====================================================================
// K1: per-batch-row fused  pool -> v=w_q@h_aspect -> scores -> softmax
//     -> opinion -> concat LayerNorm -> ln_cat (bf16)
// One workgroup per b; hidden[b] (393KB) read from HBM once, later
// passes hit L2/WGP$.
// =====================================================================
__global__ __launch_bounds__(256) void fused_pre_kernel(
    const float* __restrict__ hidden,   // [B,S,H]
    const int*   __restrict__ amask_g,  // [B,S]
    const int*   __restrict__ aspm_g,   // [B,S]
    const float* __restrict__ wq,       // [H,H]
    const float* __restrict__ gamma,    // [2H]
    const float* __restrict__ beta,     // [2H]
    unsigned short* __restrict__ ln_out)// [B,2H] bf16
{
    __shared__ float m_s[SS], am_s[SS];
    __shared__ float ha[HH], vv[HH], hop[HH], sc[SS];
    __shared__ float red[256];

    const int b = blockIdx.x;
    const int t = threadIdx.x;
    const float* hb = hidden + (size_t)b * SS * HH;

    if (t < SS) {
        m_s[t]  = (float)aspm_g[b * SS + t];
        am_s[t] = (float)amask_g[b * SS + t];
    }
    __syncthreads();

    // ---- masked-mean pool
    float cnt = block_sum((t < SS) ? m_s[t] : 0.f, red, t);
    float inv_denom = 1.0f / fmaxf(cnt, 1.0f);
    for (int h = t; h < HH; h += 256) {
        float s = 0.f;
        for (int j = 0; j < SS; ++j) s += hb[j * HH + h] * m_s[j];
        ha[h] = s * inv_denom;
    }
    __syncthreads();

    // ---- v = (w_q @ h_aspect) / sqrt(H)   (per-row GEMV, w_q L2-resident)
    const float scl = 0.03608439182435161f;  // 1/sqrt(768)
    for (int h = t; h < HH; h += 256) {
        const float* wr = wq + (size_t)h * HH;
        float s = 0.f;
        for (int k = 0; k < HH; ++k) s += wr[k] * ha[k];
        vv[h] = s * scl;
    }
    __syncthreads();

    // ---- scores[s] = hidden[b,s]·v  with masks
    if (t < SS) {
        const float* hr = hb + (size_t)t * HH;
        float s = 0.f;
        for (int k = 0; k < HH; ++k) s += hr[k] * vv[k];
        if (am_s[t] == 0.f || m_s[t] != 0.f) s = -3.4028234663852886e38f;
        sc[t] = s;
    }
    __syncthreads();

    // ---- softmax over S
    float x  = (t < SS) ? sc[t] : -3.4028234663852886e38f;
    float mx = block_max(x, red, t);
    float ex = (t < SS) ? expf(sc[t] - mx) : 0.f;
    float sm = block_sum(ex, red, t);
    if (t < SS) sc[t] = ex / sm;
    __syncthreads();

    // ---- h_opinion = attn · hidden
    for (int h = t; h < HH; h += 256) {
        float s = 0.f;
        for (int j = 0; j < SS; ++j) s += sc[j] * hb[j * HH + h];
        hop[h] = s;
    }
    __syncthreads();

    // ---- LayerNorm over concat [h_opinion, h_aspect]  (2H = 1536)
    float ls = 0.f;
    for (int i = t; i < 2 * HH; i += 256) ls += (i < HH) ? hop[i] : ha[i - HH];
    float mu = block_sum(ls, red, t) * (1.0f / (2.0f * HH));
    float lv = 0.f;
    for (int i = t; i < 2 * HH; i += 256) {
        float c = ((i < HH) ? hop[i] : ha[i - HH]) - mu;
        lv += c * c;
    }
    float var  = block_sum(lv, red, t) * (1.0f / (2.0f * HH));
    float rstd = rsqrtf(var + 1e-5f);
    unsigned short* dst = ln_out + (size_t)b * (2 * HH);
    for (int i = t; i < 2 * HH; i += 256) {
        float c = ((i < HH) ? hop[i] : ha[i - HH]) - mu;
        dst[i] = f2bf(c * rstd * gamma[i] + beta[i]);
    }
}

// =====================================================================
// Generic WMMA GEMM:  C[M,N] = A_bf16[M,K] @ bf16(B_f32[K,N])  (+ epilogue)
//   EPI 0: y = x + bias;  Cf = y; Cb = bf16(y)           (fusion linear)
//   EPI 1: y = gelu(x + bias) * rscale[row];  Cb = bf16(y)   (expert fc1)
//   EPI 2: Cf[row,col] += x                               (expert fc2 accum)
// Block tile 128x128x64, 8 waves, 4x2 WMMA tiles per wave.
// =====================================================================
#define GBM 128
#define GBN 128
#define GBK 64
#define LDA 72   // bf16 pitch: 144B rows -> 16B-aligned vec stores
#define LDB 72

template<int EPI>
__global__ __launch_bounds__(256) void gemm_bf16_kernel(
    const unsigned short* __restrict__ A,   // bf16 [M,K]
    const float* __restrict__ Bw,           // f32  [K,N]
    const float* __restrict__ bias,         // [N] (EPI 0/1)
    float* __restrict__ Cf,                 // f32 out (EPI 0 write / EPI 2 accum)
    unsigned short* __restrict__ Cb,        // bf16 out (EPI 0/1)
    const float* __restrict__ rscale, int rstride,  // EPI 1 per-row scale
    int M, int N, int K)
{
    __shared__ unsigned short As[GBM * LDA];
    __shared__ unsigned short Bs[GBN * LDB];   // n-major: Bs[n][k]

    const int tid  = threadIdx.x;
    const int lane = tid & 31;
    const int wave = tid >> 5;
    const int wm   = wave & 1;        // 2 wave-rows  (64 M each)
    const int wn   = wave >> 1;       // 4 wave-cols  (32 N each)
    const int bm   = blockIdx.y * GBM;
    const int bn   = blockIdx.x * GBN;

    v8f acc[4][2] = {};

    for (int kt = 0; kt < K; kt += GBK) {
        // ---- stage A (bf16, direct copy, 16B vectors)
        #pragma unroll
        for (int i = 0; i < 4; ++i) {
            int e = tid + i * 256;             // 128 rows x 8 chunks of 8 bf16
            int row = e >> 3, c = e & 7;
            uint4 d = *(const uint4*)(A + (size_t)(bm + row) * K + kt + c * 8);
            *(uint4*)(As + row * LDA + c * 8) = d;
        }
        // ---- stage B (f32 -> bf16, transpose into n-major LDS)
        #pragma unroll
        for (int i = 0; i < 8; ++i) {
            int e = tid + i * 256;             // 64 k-rows x 32 float4
            int k = e >> 5, nv = e & 31;
            float4 d = *(const float4*)(Bw + (size_t)(kt + k) * N + bn + nv * 4);
            Bs[(nv * 4 + 0) * LDB + k] = f2bf(d.x);
            Bs[(nv * 4 + 1) * LDB + k] = f2bf(d.y);
            Bs[(nv * 4 + 2) * LDB + k] = f2bf(d.z);
            Bs[(nv * 4 + 3) * LDB + k] = f2bf(d.w);
        }
        __syncthreads();

        #pragma unroll
        for (int ks = 0; ks < 2; ++ks) {       // two K=32 sub-steps
            Frag af[4], bf[2];
            // A frag: lane l -> row l%16, K += 8 if l>=16; v -> K = 16*(v/4)+2*(v%4)
            const int kA = ks * 32 + ((lane >> 4) << 3);
            #pragma unroll
            for (int a = 0; a < 4; ++a) {
                int row = wm * 64 + a * 16 + (lane & 15);
                const unsigned short* p = As + row * LDA + kA;
                #pragma unroll
                for (int v = 0; v < 8; ++v)
                    af[a].u[v] = *(const unsigned*)(p + ((v >> 2) << 4) + ((v & 3) << 1));
            }
            // B frag: lane l -> n = l%16, K += 8 if l>=16; v -> K = 16*(v/4)+2*(v%4)
            const int kB = ks * 32 + ((lane >> 4) << 3);
            #pragma unroll
            for (int bb = 0; bb < 2; ++bb) {
                int n = wn * 32 + bb * 16 + (lane & 15);
                const unsigned short* p = Bs + n * LDB + kB;
                #pragma unroll
                for (int v = 0; v < 8; ++v)
                    bf[bb].u[v] = *(const unsigned*)(p + ((v >> 2) << 4) + ((v & 3) << 1));
            }
            #pragma unroll
            for (int a = 0; a < 4; ++a)
                #pragma unroll
                for (int bb = 0; bb < 2; ++bb)
                    acc[a][bb] = __builtin_amdgcn_wmma_f32_16x16x32_bf16(
                        false, af[a].v, false, bf[bb].v,
                        (short)0, acc[a][bb], false, false);
        }
        __syncthreads();
    }

    // ---- epilogue: C layout VGPR j -> M = j + 8*(lane>=16), N = lane%16
    const int nb = bn + wn * 32 + (lane & 15);
    const int mb = bm + wm * 64 + ((lane >> 4) << 3);
    #pragma unroll
    for (int a = 0; a < 4; ++a) {
        #pragma unroll
        for (int bb = 0; bb < 2; ++bb) {
            int col = nb + bb * 16;
            float bv = (EPI == 2) ? 0.f : bias[col];
            #pragma unroll
            for (int j = 0; j < 8; ++j) {
                int row = mb + a * 16 + j;
                size_t off = (size_t)row * N + col;
                float x = acc[a][bb][j];
                if (EPI == 0) {
                    float y = x + bv;
                    Cf[off] = y;
                    Cb[off] = f2bf(y);
                } else if (EPI == 1) {
                    float y = x + bv;
                    float g = 0.5f * y * (1.0f + erff(y * 0.70710678118654752f));
                    g *= rscale[(size_t)row * rstride];
                    Cb[off] = f2bf(g);
                } else {
                    Cf[off] += x;
                }
            }
        }
    }
}

// =====================================================================
// K3: hierarchical router + moe init:  moe[b] = h_fused[b] + sum_ge w*b2
// One wave per batch row.
// =====================================================================
__global__ __launch_bounds__(256) void router_kernel(
    const float* __restrict__ hf,    // [B,H] h_fused f32
    const float* __restrict__ w_gr,  // [H,G]
    const float* __restrict__ b_gr,  // [G]
    const float* __restrict__ w_er,  // [G,H,E]
    const float* __restrict__ b_er,  // [G,E]
    const float* __restrict__ b2,    // [G,E,H]
    float* __restrict__ rw,          // [B,24]
    float* __restrict__ moe)         // [B,H]
{
    __shared__ float lrw[8][NEXP];
    const int wave = threadIdx.x >> 5;
    const int lane = threadIdx.x & 31;
    const int b = blockIdx.x * 8 + wave;
    const float* h = hf + (size_t)b * HH;

    float gl[GG] = {0.f, 0.f, 0.f};
    float el[NEXP];
    #pragma unroll
    for (int j = 0; j < NEXP; ++j) el[j] = 0.f;

    for (int k = lane; k < HH; k += 32) {
        float x = h[k];
        #pragma unroll
        for (int j = 0; j < GG; ++j) gl[j] += x * w_gr[k * GG + j];
        #pragma unroll
        for (int g = 0; g < GG; ++g)
            #pragma unroll
            for (int e = 0; e < EE; ++e)
                el[g * EE + e] += x * w_er[((size_t)g * HH + k) * EE + e];
    }
    #pragma unroll
    for (int off = 16; off > 0; off >>= 1) {
        #pragma unroll
        for (int j = 0; j < GG; ++j)   gl[j] += __shfl_down(gl[j], off, 32);
        #pragma unroll
        for (int j = 0; j < NEXP; ++j) el[j] += __shfl_down(el[j], off, 32);
    }
    if (lane == 0) {
        float g0 = gl[0] + b_gr[0], g1 = gl[1] + b_gr[1], g2 = gl[2] + b_gr[2];
        float gm = fmaxf(g0, fmaxf(g1, g2));
        float e0 = expf(g0 - gm), e1 = expf(g1 - gm), e2 = expf(g2 - gm);
        float gs = 1.0f / (e0 + e1 + e2);
        float gp[GG] = {e0 * gs, e1 * gs, e2 * gs};
        #pragma unroll
        for (int g = 0; g < GG; ++g) {
            float l[EE], m = -3.4e38f;
            #pragma unroll
            for (int e = 0; e < EE; ++e) { l[e] = el[g * EE + e] + b_er[g * EE + e]; m = fmaxf(m, l[e]); }
            float s = 0.f;
            #pragma unroll
            for (int e = 0; e < EE; ++e) { l[e] = expf(l[e] - m); s += l[e]; }
            float inv = 1.0f / s;
            #pragma unroll
            for (int e = 0; e < EE; ++e) {
                float wv = gp[g] * l[e] * inv;
                lrw[wave][g * EE + e] = wv;
                rw[(size_t)b * NEXP + g * EE + e] = wv;
            }
        }
    }
    __syncthreads();
    // moe init = residual h_fused + sum_ge w * b2
    for (int hh = lane; hh < HH; hh += 32) {
        float acc = h[hh];
        #pragma unroll
        for (int j = 0; j < NEXP; ++j) acc += lrw[wave][j] * b2[(size_t)j * HH + hh];
        moe[(size_t)b * HH + hh] = acc;
    }
}

// =====================================================================
// K6: classifier head  logits[b,l] = (h_fused+moe)[b]·w_cls[:,l] + b_cls[l]
// =====================================================================
__global__ __launch_bounds__(256) void cls_kernel(
    const float* __restrict__ hout,   // [B,H] (moe buffer after accumulation)
    const float* __restrict__ w_cls,  // [H,L]
    const float* __restrict__ b_cls,  // [L]
    float* __restrict__ out)          // [B,L]
{
    int idx = blockIdx.x * 256 + threadIdx.x;   // B*L = 6144
    int b = idx / LL, l = idx - b * LL;
    const float* h = hout + (size_t)b * HH;
    float acc = b_cls[l];
    for (int k = 0; k < HH; ++k) acc += h[k] * w_cls[k * LL + l];
    out[idx] = acc;
}

// =====================================================================
extern "C" void kernel_launch(void* const* d_in, const int* in_sizes, int n_in,
                              void* d_out, int out_size, void* d_ws, size_t ws_size,
                              hipStream_t stream) {
    const float* hidden = (const float*)d_in[0];
    const int*   amask  = (const int*)  d_in[1];
    const int*   aspm   = (const int*)  d_in[2];
    const float* w_q    = (const float*)d_in[3];
    const float* gamma  = (const float*)d_in[4];
    const float* beta   = (const float*)d_in[5];
    const float* w_fc   = (const float*)d_in[6];
    const float* b_fc   = (const float*)d_in[7];
    const float* w_gr   = (const float*)d_in[8];
    const float* b_gr   = (const float*)d_in[9];
    const float* w_er   = (const float*)d_in[10];
    const float* b_er   = (const float*)d_in[11];
    const float* w1     = (const float*)d_in[12];
    const float* b1     = (const float*)d_in[13];
    const float* w2     = (const float*)d_in[14];
    const float* b2     = (const float*)d_in[15];
    const float* w_cls  = (const float*)d_in[16];
    const float* b_cls  = (const float*)d_in[17];
    float* out = (float*)d_out;

    // workspace layout (~34.8 MB)
    char* w = (char*)d_ws;
    unsigned short* ln_cat  = (unsigned short*)(w + 0);           // [B,1536] bf16
    float*          h_fused = (float*)         (w + 6291456);     // [B,768]  f32
    unsigned short* hf_bf   = (unsigned short*)(w + 12582912);    // [B,768]  bf16
    float*          rw      = (float*)         (w + 15728640);    // [B,24]   f32
    float*          moe     = (float*)         (w + 15925248);    // [B,768]  f32
    unsigned short* h1      = (unsigned short*)(w + 22216704);    // [B,3072] bf16

    // 1) fused pool / attention / LayerNorm  (single HBM pass over hidden)
    fused_pre_kernel<<<BB, 256, 0, stream>>>(hidden, amask, aspm, w_q, gamma, beta, ln_cat);

    // 2) h_fused = ln_cat @ w_fc + b_fc      [2048,1536]x[1536,768]
    gemm_bf16_kernel<0><<<dim3(HH / GBN, BB / GBM), 256, 0, stream>>>(
        ln_cat, w_fc, b_fc, h_fused, hf_bf, nullptr, 0, BB, HH, 2 * HH);

    // 3) router weights + moe init (residual + weighted b2)
    router_kernel<<<BB / 8, 256, 0, stream>>>(h_fused, w_gr, b_gr, w_er, b_er, b2, rw, moe);

    // 4) experts: fc1 (GELU * route-weight, bf16) then fc2 accumulate into moe
    for (int ge = 0; ge < NEXP; ++ge) {
        gemm_bf16_kernel<1><<<dim3(II / GBN, BB / GBM), 256, 0, stream>>>(
            hf_bf, w1 + (size_t)ge * HH * II, b1 + (size_t)ge * II,
            nullptr, h1, rw + ge, NEXP, BB, II, HH);
        gemm_bf16_kernel<2><<<dim3(HH / GBN, BB / GBM), 256, 0, stream>>>(
            h1, w2 + (size_t)ge * II * HH, nullptr,
            moe, nullptr, nullptr, 0, BB, HH, II);
    }

    // 5) classifier head
    cls_kernel<<<(BB * LL) / 256, 256, 0, stream>>>(moe, w_cls, b_cls, out);
}
